// CVAE_78881369358606
// MI455X (gfx1250) — matrix operand order repaired and checked by reading, not compiled, not run on hardware
//
#include <hip/hip_runtime.h>

// ---------------------------------------------------------------------------
// Problem constants (from the reference)
// ---------------------------------------------------------------------------
#define NV   5023
#define NE   30138
#define NB   64
#define NLAT 32
#define NTOT (NV * 32)          // decoder GEMM N dimension = 160736
#define BN_EPS 1e-3f

typedef __attribute__((ext_vector_type(16))) __bf16 v16bf;
typedef __attribute__((ext_vector_type(8)))  float  v8f;

// ---------------------------------------------------------------------------
// CSR row pointers via binary search on sorted edge_row
// ---------------------------------------------------------------------------
__global__ void rowptr_kernel(const int* __restrict__ erow, int* __restrict__ rowptr) {
    int v = blockIdx.x * blockDim.x + threadIdx.x;
    if (v > NV) return;
    int lo = 0, hi = NE;
    while (lo < hi) {
        int mid = (lo + hi) >> 1;
        if (erow[mid] < v) lo = mid + 1; else hi = mid;
    }
    rowptr[v] = lo;
}

// ---------------------------------------------------------------------------
// Encoder vertex MLP: h1 = relu(bn1(x@W0+b0)), xu = h1@ue   (3->16->8)
// ---------------------------------------------------------------------------
__global__ __launch_bounds__(256) void enc_kernel(
    const float* __restrict__ x, const float* __restrict__ W0, const float* __restrict__ b0,
    const float* __restrict__ g, const float* __restrict__ bt,
    const float* __restrict__ mu, const float* __restrict__ var,
    const float* __restrict__ ue,
    float* __restrict__ h1, float* __restrict__ xu)
{
    size_t i = (size_t)blockIdx.x * blockDim.x + threadIdx.x;   // (b,v) flat
    if (i >= (size_t)NB * NV) return;
    float x0 = x[i * 3 + 0], x1 = x[i * 3 + 1], x2 = x[i * 3 + 2];
    float h[16];
    #pragma unroll
    for (int j = 0; j < 16; ++j) {
        float a = b0[j] + x0 * W0[j] + x1 * W0[16 + j] + x2 * W0[32 + j];
        float sc = g[j] * rsqrtf(var[j] + BN_EPS);
        a = (a - mu[j]) * sc + bt[j];
        h[j] = fmaxf(a, 0.f);
        h1[i * 16 + j] = h[j];
    }
    #pragma unroll
    for (int m = 0; m < 8; ++m) {
        float s = 0.f;
        #pragma unroll
        for (int j = 0; j < 16; ++j) s += h[j] * ue[j * 8 + m];
        xu[i * 8 + m] = s;
    }
}

// xu = h @ u for D=32 feature rows (decoder-side attention projections)
__global__ __launch_bounds__(256) void xu_kernel(
    const float* __restrict__ h, const float* __restrict__ u, float* __restrict__ xu)
{
    size_t i = (size_t)blockIdx.x * blockDim.x + threadIdx.x;
    if (i >= (size_t)NB * NV) return;
    const float* hr = &h[i * 32];
    #pragma unroll
    for (int m = 0; m < 8; ++m) {
        float s = 0.f;
        #pragma unroll
        for (int d = 0; d < 32; ++d) s += hr[d] * u[d * 8 + m];
        xu[i * 8 + m] = s;
    }
}

// ---------------------------------------------------------------------------
// FeaST convolution, tile of 16 vertices per workgroup.
// Phase 1: per-edge softmax(8 heads) + aggregation into LDS agg[16][8*DIN]
//          (8 threads per vertex, one head each -> no write conflicts)
// Phase 2: WMMA  agg(16 x 8*DIN) @ W(8*DIN x DOUT)  in bf16, f32 accum,
//          fused bias (+ BN) + ReLU epilogue.
// ---------------------------------------------------------------------------
template<int DIN, int DOUT, bool BN>
__global__ __launch_bounds__(128) void feast_kernel(
    const float* __restrict__ hin, const float* __restrict__ xu,
    const int* __restrict__ rowptr, const int* __restrict__ ecol,
    const float* __restrict__ ew, const float* __restrict__ W,
    const float* __restrict__ cvec, const float* __restrict__ bias,
    const float* __restrict__ bng, const float* __restrict__ bnb,
    const float* __restrict__ bnmu, const float* __restrict__ bnvar,
    float* __restrict__ hout)
{
    constexpr int K = 8 * DIN;
    __shared__ float aggf[16][K];
    __shared__ float xur[16][8];
    __shared__ float cs[8];

    const int b  = blockIdx.y;
    const int v0 = blockIdx.x * 16;
    const int t  = threadIdx.x;

    if (t < 8) cs[t] = cvec[t];
    {   // target-vertex xu rows
        int vl = t >> 3, m = t & 7;
        int v = v0 + vl;
        xur[vl][m] = (v < NV) ? xu[((size_t)b * NV + v) * 8 + m] : 0.f;
    }
    for (int i = t; i < 16 * K; i += 128) (&aggf[0][0])[i] = 0.f;
    __syncthreads();

    {   // per-edge softmax + weighted gather
        int g = t >> 3;   // local vertex 0..15
        int j = t & 7;    // head handled by this thread
        int v = v0 + g;
        if (v < NV) {
            int e0 = rowptr[v], e1 = rowptr[v + 1];
            for (int e = e0; e < e1; ++e) {
                int   col = ecol[e];
                float a   = ew[e];
                const float* xc = &xu[((size_t)b * NV + col) * 8];
                float l[8], mx = -1e30f;
                #pragma unroll
                for (int m = 0; m < 8; ++m) { l[m] = xc[m] - xur[g][m] + cs[m]; mx = fmaxf(mx, l[m]); }
                float s = 0.f;
                #pragma unroll
                for (int m = 0; m < 8; ++m) { l[m] = __expf(l[m] - mx); s += l[m]; }
                float q = a * l[j] / s;
                const float* hr  = &hin[((size_t)b * NV + col) * DIN];
                float*       dst = &aggf[g][j * DIN];
                #pragma unroll
                for (int d = 0; d < DIN; ++d) dst[d] += q * hr[d];
            }
        }
    }
    __syncthreads();

    // ---- WMMA phase: M=16 vertices, N=DOUT, K=8*DIN ----
    const int wave = t >> 5;
    const int lane = t & 31;
    constexpr int NTILES = DOUT / 16;
    if (wave < NTILES) {
        const int ncol  = wave * 16 + (lane & 15);
        const int mrow  = lane & 15;
        const int koffA = (lane < 16) ? 0 : 16;   // A: contiguous K-half by lane group
        const int koffB = (lane < 16) ? 0 : 16;   // B: lane = N, K-half by lane group
        v8f acc = {};
        for (int kc = 0; kc < K; kc += 32) {
            v16bf af, bfv;
            #pragma unroll
            for (int i = 0; i < 8; ++i) {
                af[i]     = (__bf16)aggf[mrow][kc + koffA + i];
                af[i + 8] = (__bf16)aggf[mrow][kc + 8 + koffA + i];
            }
            #pragma unroll
            for (int i = 0; i < 16; ++i)
                bfv[i] = (__bf16)W[(size_t)(kc + koffB + i) * DOUT + ncol];
            acc = __builtin_amdgcn_wmma_f32_16x16x32_bf16(
                      false, af, false, bfv, (short)0, acc, false, false);
        }
        // fused epilogue: bias (+BN) + ReLU
        float bia = bias[ncol];
        float scale = 1.f, shift = 0.f;
        if (BN) {
            scale = bng[ncol] * rsqrtf(bnvar[ncol] + BN_EPS);
            shift = bnb[ncol] - bnmu[ncol] * scale;
        }
        #pragma unroll
        for (int r = 0; r < 8; ++r) {
            int m = r + ((lane >= 16) ? 8 : 0);
            int v = v0 + m;
            if (v < NV) {
                float val = acc[r] + bia;
                if (BN) val = val * scale + shift;
                val = fmaxf(val, 0.f);
                hout[((size_t)b * NV + v) * DOUT + ncol] = val;
            }
        }
    }
}

// ---------------------------------------------------------------------------
// Mean-pool over V: (B,V,32) -> (B,32)
// ---------------------------------------------------------------------------
__global__ __launch_bounds__(256) void meanpool_kernel(
    const float* __restrict__ h, float* __restrict__ mean)
{
    __shared__ float red[8][32];
    int b = blockIdx.x;
    int c = threadIdx.x & 31, sub = threadIdx.x >> 5;
    float s = 0.f;
    for (int v = sub; v < NV; v += 8) s += h[((size_t)b * NV + v) * 32 + c];
    red[sub][c] = s;
    __syncthreads();
    if (threadIdx.x < 32) {
        float tot = 0.f;
        #pragma unroll
        for (int i = 0; i < 8; ++i) tot += red[i][c];
        mean[b * 32 + c] = tot / (float)NV;
    }
}

// stats = mean@Wfc+bfc ; z = eps*exp(0.5*logvar)+mu      (one tiny block)
__global__ void latent_kernel(const float* __restrict__ mean,
                              const float* __restrict__ Wfc, const float* __restrict__ bfc,
                              const float* __restrict__ eps, float* __restrict__ z)
{
    int b = threadIdx.x;       // 64 threads
    if (b >= NB) return;
    float m[32];
    #pragma unroll
    for (int k = 0; k < 32; ++k) m[k] = mean[b * 32 + k];
    for (int l = 0; l < 32; ++l) {
        float mu = bfc[l], lv = bfc[32 + l];
        #pragma unroll
        for (int k = 0; k < 32; ++k) {
            mu += m[k] * Wfc[k * 64 + l];
            lv += m[k] * Wfc[k * 64 + 32 + l];
        }
        z[b * 32 + l] = eps[b * 32 + l] * __expf(0.5f * lv) + mu;
    }
}

// ---------------------------------------------------------------------------
// Decoder GEMM: h3 = relu(bn3(z(64x32) @ Wd(32 x V*32) + bd))
// 256 threads / WG; LDS-staged bf16 Wd tile (32x128) + full z; WMMA.
// ---------------------------------------------------------------------------
__global__ __launch_bounds__(256) void dec_gemm_kernel(
    const float* __restrict__ z, const float* __restrict__ Wd,
    const float* __restrict__ bd,
    const float* __restrict__ bng, const float* __restrict__ bnb,
    const float* __restrict__ bnmu, const float* __restrict__ bnvar,
    float* __restrict__ hout)
{
    __shared__ __bf16 zs[64][32];
    __shared__ __bf16 wt[32][128];

    const int t     = threadIdx.x;
    const int nbase = blockIdx.x * 128;

    for (int i = t; i < 64 * 32; i += 256) (&zs[0][0])[i] = (__bf16)z[i];
    for (int i = t; i < 32 * 128; i += 256) {
        int k = i >> 7, nl = i & 127;
        int n = nbase + nl;
        wt[k][nl] = (n < NTOT) ? (__bf16)Wd[(size_t)k * NTOT + n] : (__bf16)0.f;
        // speculative prefetch of next workgroup's tile (global_prefetch_b8)
        if (n + 128 < NTOT) __builtin_prefetch(&Wd[(size_t)k * NTOT + n + 128], 0, 1);
    }
    __syncthreads();

    const int wave = t >> 5, lane = t & 31;
    const int nl = wave * 16 + (lane & 15);
    const int n  = nbase + nl;
    const int koffA = (lane < 16) ? 0 : 16;
    const int koffB = (lane < 16) ? 0 : 16;

    v16bf bfv;
    #pragma unroll
    for (int i = 0; i < 16; ++i) bfv[i] = wt[koffB + i][nl];

    const int ch = n & 31;                       // feature channel = n % 32
    float scale = bng[ch] * rsqrtf(bnvar[ch] + BN_EPS);
    float shift = bnb[ch] - bnmu[ch] * scale;
    float bia   = (n < NTOT) ? bd[n] : 0.f;

    #pragma unroll
    for (int mt = 0; mt < 4; ++mt) {             // 64 batch rows = 4 M-tiles
        v16bf af;
        int mrow = mt * 16 + (lane & 15);
        #pragma unroll
        for (int i = 0; i < 8; ++i) {
            af[i]     = zs[mrow][koffA + i];
            af[i + 8] = zs[mrow][8 + koffA + i];
        }
        v8f acc = {};
        acc = __builtin_amdgcn_wmma_f32_16x16x32_bf16(
                  false, af, false, bfv, (short)0, acc, false, false);
        if (n < NTOT) {
            #pragma unroll
            for (int r = 0; r < 8; ++r) {
                int bb = mt * 16 + r + ((lane >= 16) ? 8 : 0);
                float val = acc[r] + bia;
                val = fmaxf(val * scale + shift, 0.f);
                hout[(size_t)bb * NTOT + n] = val;
            }
        }
    }
}

// ---------------------------------------------------------------------------
// Output head: out = h5(B,V,16) @ Wout(16,3) + bout
// ---------------------------------------------------------------------------
__global__ __launch_bounds__(256) void out_kernel(
    const float* __restrict__ h, const float* __restrict__ Wout,
    const float* __restrict__ bout, float* __restrict__ out)
{
    size_t i = (size_t)blockIdx.x * blockDim.x + threadIdx.x;
    if (i >= (size_t)NB * NV) return;
    const float* hr = &h[i * 16];
    #pragma unroll
    for (int j = 0; j < 3; ++j) {
        float s = bout[j];
        #pragma unroll
        for (int d = 0; d < 16; ++d) s += hr[d] * Wout[d * 3 + j];
        out[i * 3 + j] = s;
    }
}

// ---------------------------------------------------------------------------
// Host launcher
// ---------------------------------------------------------------------------
extern "C" void kernel_launch(void* const* d_in, const int* in_sizes, int n_in,
                              void* d_out, int out_size, void* d_ws, size_t ws_size,
                              hipStream_t stream) {
    (void)in_sizes; (void)n_in; (void)out_size; (void)ws_size;

    // ---- inputs (setup_inputs dict order; params flattened in-order) ----
    const float* x        = (const float*)d_in[0];
    const float* eps      = (const float*)d_in[1];
    const int*   edge_row = (const int*)  d_in[2];
    const int*   edge_col = (const int*)  d_in[3];
    const float* edge_w   = (const float*)d_in[4];
    const float* W0   = (const float*)d_in[5];
    const float* b0   = (const float*)d_in[6];
    const float* bn1g = (const float*)d_in[7];
    const float* bn1b = (const float*)d_in[8];
    const float* bn1m = (const float*)d_in[9];
    const float* bn1v = (const float*)d_in[10];
    const float* ue   = (const float*)d_in[11];
    const float* ce   = (const float*)d_in[12];
    const float* we   = (const float*)d_in[13];
    const float* be   = (const float*)d_in[14];
    const float* bn2g = (const float*)d_in[15];
    const float* bn2b = (const float*)d_in[16];
    const float* bn2m = (const float*)d_in[17];
    const float* bn2v = (const float*)d_in[18];
    const float* Wfc  = (const float*)d_in[19];
    const float* bfc  = (const float*)d_in[20];
    const float* Wd   = (const float*)d_in[21];
    const float* bd   = (const float*)d_in[22];
    const float* bn3g = (const float*)d_in[23];
    const float* bn3b = (const float*)d_in[24];
    const float* bn3m = (const float*)d_in[25];
    const float* bn3v = (const float*)d_in[26];
    const float* ud1  = (const float*)d_in[27];
    const float* cd1  = (const float*)d_in[28];
    const float* wd1  = (const float*)d_in[29];
    const float* bd1  = (const float*)d_in[30];
    const float* bn4g = (const float*)d_in[31];
    const float* bn4b = (const float*)d_in[32];
    const float* bn4m = (const float*)d_in[33];
    const float* bn4v = (const float*)d_in[34];
    const float* ud2  = (const float*)d_in[35];
    const float* cd2  = (const float*)d_in[36];
    const float* wd2  = (const float*)d_in[37];
    const float* bd2  = (const float*)d_in[38];
    const float* Wout = (const float*)d_in[39];
    const float* bout = (const float*)d_in[40];

    // ---- workspace carve-up (256B aligned) ----
    char* ws = (char*)d_ws;
    size_t off = 0;
    auto carve = [&](size_t bytes) -> char* {
        char* p = ws + off;
        off += (bytes + 255) & ~(size_t)255;
        return p;
    };
    int*   rowptr = (int*)  carve((NV + 1) * sizeof(int));
    float* h16    = (float*)carve((size_t)NB * NV * 16 * sizeof(float)); // h1, later h5
    float* xu     = (float*)carve((size_t)NB * NV * 8  * sizeof(float));
    float* h32a   = (float*)carve((size_t)NB * NV * 32 * sizeof(float)); // h2, later h4
    float* h32b   = (float*)carve((size_t)NB * NV * 32 * sizeof(float)); // h3
    float* meanb  = (float*)carve(NB * 32 * sizeof(float));
    float* zbuf   = (float*)carve(NB * 32 * sizeof(float));

    const size_t NBV = (size_t)NB * NV;
    const int vtiles = (NV + 15) / 16;
    dim3 feast_grid(vtiles, NB);

    // 1. CSR row pointers
    rowptr_kernel<<<(NV + 1 + 255) / 256, 256, 0, stream>>>(edge_row, rowptr);

    // 2. encoder vertex MLP + attention projection
    enc_kernel<<<(NBV + 255) / 256, 256, 0, stream>>>(
        x, W0, b0, bn1g, bn1b, bn1m, bn1v, ue, h16, xu);

    // 3. FeaST conv 1 (16 -> 32) + bn2 + relu
    feast_kernel<16, 32, true><<<feast_grid, 128, 0, stream>>>(
        h16, xu, rowptr, edge_col, edge_w, we, ce, be,
        bn2g, bn2b, bn2m, bn2v, h32a);

    // 4. mean pool + latent reparameterization
    meanpool_kernel<<<NB, 256, 0, stream>>>(h32a, meanb);
    latent_kernel<<<1, 64, 0, stream>>>(meanb, Wfc, bfc, eps, zbuf);

    // 5. decoder GEMM (WMMA) + bn3 + relu
    dec_gemm_kernel<<<(NTOT + 127) / 128, 256, 0, stream>>>(
        zbuf, Wd, bd, bn3g, bn3b, bn3m, bn3v, h32b);

    // 6. FeaST conv 2 (32 -> 32) + bn4 + relu
    xu_kernel<<<(NBV + 255) / 256, 256, 0, stream>>>(h32b, ud1, xu);
    feast_kernel<32, 32, true><<<feast_grid, 128, 0, stream>>>(
        h32b, xu, rowptr, edge_col, edge_w, wd1, cd1, bd1,
        bn4g, bn4b, bn4m, bn4v, h32a);

    // 7. FeaST conv 3 (32 -> 16) + relu (no BN)
    xu_kernel<<<(NBV + 255) / 256, 256, 0, stream>>>(h32a, ud2, xu);
    feast_kernel<32, 16, false><<<feast_grid, 128, 0, stream>>>(
        h32a, xu, rowptr, edge_col, edge_w, wd2, cd2, bd2,
        nullptr, nullptr, nullptr, nullptr, h16);

    // 8. output head
    out_kernel<<<(NBV + 255) / 256, 256, 0, stream>>>(
        h16, Wout, bout, (float*)d_out);
}
